// CustomModel_22204980920786
// MI455X (gfx1250) — matrix operand order repaired and checked by reading, not compiled, not run on hardware
//
#include <hip/hip_runtime.h>
#include <hip/hip_bf16.h>
#include <math.h>

typedef _Float16 f16;
typedef __attribute__((ext_vector_type(16))) _Float16 v16h;
typedef __attribute__((ext_vector_type(8)))  _Float16 v8h;
typedef __attribute__((ext_vector_type(8)))  float    v8f;

#define BATCH 2048
#define HW    81
#define NPIX  (BATCH*HW)     // 165888
#define CINC  39
#define K1    352            // 351 padded to 11*32
#define K1V   351
#define KC1   11
#define K2    576            // 64ch * 9 taps, 18*32
#define KC2   18
#define HOFF  (1 + BATCH*18*HW)   // start of h in d_out

// workspace byte offsets
#define WS_SUMS 0                        // 256 floats: sum1,sq1,sum2,sq2
#define WS_WF1  1024                     // 22528 halves
#define WS_WF2  (WS_WF1 + 22528*2)       // 18432 halves
#define WS_WF3  (WS_WF2 + 18432*2)       // 36864 halves
#define WS_H1   156672                   // NPIX*64 f32 (relu/bn'd conv1)
#define WS_MASK (WS_H1 + (size_t)NPIX*64*4)
#define WS_H2   (WS_MASK + (size_t)NPIX*9*4)

// ---------------- prep: swizzle weights into WMMA B-fragment order ----------
__global__ void prep_kernel(const float* __restrict__ w1, const float* __restrict__ wo,
                            const float* __restrict__ wm, const float* __restrict__ wd,
                            float* __restrict__ out, float* __restrict__ sums,
                            f16* __restrict__ wf1, f16* __restrict__ wf2, f16* __restrict__ wf3) {
  int id = blockIdx.x * 256 + threadIdx.x;
  if (id == 0) out[0] = 0.5f;                 // prob
  if (id < 256) sums[id] = 0.0f;              // BN accumulators
  const int F1 = KC1 * 4 * 512;               // 22528
  const int F2 = KC2 * 2 * 512;               // 18432
  const int F3 = KC2 * 4 * 512;               // 36864
  if (id < F1) {
    int e = id & 15, lane = (id >> 4) & 31, nT = (id >> 9) & 3, kc = id >> 11;
    int k = kc * 32 + (e < 8 ? 8 * (lane >> 4) + e : 16 + 8 * (lane >> 4) + (e - 8));       // B frag: K = kc*32 + (e<8 ? 8g+e : 16+8g+(e-8))
    int n = nT * 16 + (lane & 15);
    float v = 0.f;
    if (k < K1V) { int t = k / CINC, c = k % CINC; v = w1[(n * CINC + c) * 9 + t]; }
    wf1[id] = (f16)v;
  } else if (id < F1 + F2) {
    int lo = id - F1;
    int e = lo & 15, lane = (lo >> 4) & 31, nT = (lo >> 9) & 1, kc = lo >> 10;
    int k = kc * 32 + (e < 8 ? 8 * (lane >> 4) + e : 16 + 8 * (lane >> 4) + (e - 8));
    int t = k >> 6, c = k & 63;
    int n = nT * 16 + (lane & 15);
    float v = 0.f;
    if (n < 18)      v = wo[(n * 64 + c) * 9 + t];
    else if (n < 27) v = wm[((n - 18) * 64 + c) * 9 + t];
    wf2[lo] = (f16)v;
  } else if (id < F1 + F2 + F3) {
    int lo = id - F1 - F2;
    int e = lo & 15, lane = (lo >> 4) & 31, nT = (lo >> 9) & 3, kc = lo >> 11;
    int k = kc * 32 + (e < 8 ? 8 * (lane >> 4) + e : 16 + 8 * (lane >> 4) + (e - 8));
    int kk = k >> 6, c = k & 63;
    int n = nT * 16 + (lane & 15);
    wf3[lo] = (f16)wd[(n * 64 + c) * 9 + kk];
  }
}

// A-fragment load from an LDS row: K = kBase + (e<8 ? 8g+e : 8g+16+(e-8))
__device__ __forceinline__ v16h load_a(const f16* rowp) {
  v8h lo = *(const v8h*)rowp;
  v8h hi = *(const v8h*)(rowp + 16);
  v16h a;
#pragma unroll
  for (int i = 0; i < 8; ++i) { a[i] = lo[i]; a[8 + i] = hi[i]; }
  return a;
}

// ---------------- GEMM1: conv1 + bias + relu + BN stats ---------------------
__global__ void __launch_bounds__(256) conv1_kernel(const float* __restrict__ x,
    const float* __restrict__ b1, const f16* __restrict__ wf1,
    float* __restrict__ h1, float* __restrict__ sums) {
  __shared__ __align__(16) f16 As[32][K1];
  __shared__ float sSum[64], sSq[64];
  const int tid = threadIdx.x;
  if (tid < 64) { sSum[tid] = 0.f; sSq[tid] = 0.f; }
  const int pBase = blockIdx.x * 32;
  for (int it = tid; it < 32 * 9; it += 256) {
    int pl = it / 9, t = it % 9;
    int p = pBase + pl;
    int b = p / HW, pix = p % HW, y = pix / 9, xx = pix % 9;
    int sy = y + t / 3 - 1, sx = xx + t % 3 - 1;
    f16* dst = &As[pl][t * CINC];
    if (sy >= 0 && sy < 9 && sx >= 0 && sx < 9) {
      const float* src = x + (size_t)(b * 81 + sy * 9 + sx) * CINC;
      for (int c = 0; c < CINC; ++c) dst[c] = (f16)src[c];
    } else {
      for (int c = 0; c < CINC; ++c) dst[c] = (f16)0.f;
    }
    if (t == 0) As[pl][K1V] = (f16)0.f;   // K padding
  }
  __syncthreads();
  const int wave = tid >> 5, lane = tid & 31;
  const int mSub = wave & 1, nT = wave >> 1;
  const int g = lane >> 4, l16 = lane & 15;
  v8f acc = {};
  for (int kc = 0; kc < KC1; ++kc) {
    v16h a = load_a(&As[mSub * 16 + l16][kc * 32 + 8 * g]);
    v16h bm = *(const v16h*)(wf1 + ((size_t)(kc * 4 + nT) * 32 + lane) * 16);
    acc = __builtin_amdgcn_wmma_f32_16x16x32_f16(false, a, false, bm, (short)0, acc, false, false);
  }
  const int n = nT * 16 + l16;
  const float bias = b1[n];
  const int mrow = pBase + mSub * 16 + 8 * g;
  float ls = 0.f, lq = 0.f;
#pragma unroll
  for (int r = 0; r < 8; ++r) {
    float v = acc[r] + bias;
    v = v > 0.f ? v : 0.f;
    h1[(size_t)(mrow + r) * 64 + n] = v;
    ls += v; lq += v * v;
  }
  atomicAdd(&sSum[n], ls);
  atomicAdd(&sSq[n], lq);
  __syncthreads();
  if (tid < 64) { atomicAdd(&sums[tid], sSum[tid]); atomicAdd(&sums[64 + tid], sSq[tid]); }
}

// ---------------- BN1 normalize (in place) ----------------------------------
__global__ void bn1_kernel(const float* __restrict__ sums, const float* __restrict__ g1,
                           const float* __restrict__ be1, float* __restrict__ h1) {
  size_t idx = (size_t)blockIdx.x * 256 + threadIdx.x;
  if (idx >= (size_t)NPIX * 64) return;
  int c = (int)(idx & 63);
  const float inv = 1.0f / (float)NPIX;
  float mean = sums[c] * inv;
  float var = sums[64 + c] * inv - mean * mean;
  float s = rsqrtf(var + 1e-5f) * g1[c];
  h1[idx] = (h1[idx] - mean) * s + be1[c];
}

// ---------------- GEMM2: offset (18ch) + mask (9ch, sigmoid) ----------------
__global__ void __launch_bounds__(128) offmask_kernel(const float* __restrict__ hn,
    const float* __restrict__ bo, const float* __restrict__ bmv, const f16* __restrict__ wf2,
    float* __restrict__ out, float* __restrict__ maskbuf) {
  __shared__ __align__(16) f16 As[32][K2];
  const int tid = threadIdx.x;
  const int pBase = blockIdx.x * 32;
  for (int it = tid; it < 32 * 9; it += 128) {
    int pl = it / 9, t = it % 9;
    int p = pBase + pl;
    int b = p / HW, pix = p % HW, y = pix / 9, xx = pix % 9;
    int sy = y + t / 3 - 1, sx = xx + t % 3 - 1;
    f16* dst = &As[pl][t * 64];
    if (sy >= 0 && sy < 9 && sx >= 0 && sx < 9) {
      const float* src = hn + (size_t)(b * 81 + sy * 9 + sx) * 64;
      for (int c = 0; c < 64; ++c) dst[c] = (f16)src[c];
    } else {
      for (int c = 0; c < 64; ++c) dst[c] = (f16)0.f;
    }
  }
  __syncthreads();
  const int wave = tid >> 5, lane = tid & 31;
  const int mSub = wave >> 1, nT = wave & 1;
  const int g = lane >> 4, l16 = lane & 15;
  v8f acc = {};
  for (int kc = 0; kc < KC2; ++kc) {
    v16h a = load_a(&As[mSub * 16 + l16][kc * 32 + 8 * g]);
    v16h bm = *(const v16h*)(wf2 + ((size_t)(kc * 2 + nT) * 32 + lane) * 16);
    acc = __builtin_amdgcn_wmma_f32_16x16x32_f16(false, a, false, bm, (short)0, acc, false, false);
  }
  const int n = nT * 16 + l16;
#pragma unroll
  for (int r = 0; r < 8; ++r) {
    int p = pBase + mSub * 16 + 8 * g + r;
    int b = p / HW, pix = p % HW;
    float v = acc[r];
    if (n < 18) {
      out[1 + (size_t)(b * 18 + n) * 81 + pix] = v + bo[n];
    } else if (n < 27) {
      float s = v + bmv[n - 18];
      maskbuf[(size_t)p * 9 + (n - 18)] = 1.0f / (1.0f + expf(-s));
    }
  }
}

// ---------------- GEMM3: deformable sample + einsum + BN stats --------------
__global__ void __launch_bounds__(256) deform_kernel(const float* __restrict__ hn,
    const float* __restrict__ offs, const float* __restrict__ maskbuf,
    const float* __restrict__ bd, const f16* __restrict__ wf3,
    float* __restrict__ h2, float* __restrict__ sums) {
  __shared__ __align__(16) f16 Vs[32][K2];
  __shared__ float sSum[64], sSq[64];
  const int tid = threadIdx.x;
  if (tid < 64) { sSum[tid] = 0.f; sSq[tid] = 0.f; }
  const int pBase = blockIdx.x * 32;
  for (int it = tid; it < 32 * 9; it += 256) {
    int pl = it / 9, kk = it % 9;
    int p = pBase + pl;
    int b = p / HW, pix = p % HW, y = pix / 9, xx = pix % 9;
    float dy = offs[1 + (size_t)(b * 18 + 2 * kk) * 81 + pix];
    float dx = offs[1 + (size_t)(b * 18 + 2 * kk + 1) * 81 + pix];
    float m = maskbuf[(size_t)p * 9 + kk];
    float py = dy + (float)(y + kk / 3 - 1);
    float px = dx + (float)(xx + kk % 3 - 1);
    float y0f = floorf(py), x0f = floorf(px);
    float ly = py - y0f, lx = px - x0f;
    int y0 = (int)y0f, x0 = (int)x0f;
    int y1 = y0 + 1, x1 = x0 + 1;
    float vy0 = (y0 >= 0 && y0 < 9) ? 1.f : 0.f;
    float vy1 = (y1 >= 0 && y1 < 9) ? 1.f : 0.f;
    float vx0 = (x0 >= 0 && x0 < 9) ? 1.f : 0.f;
    float vx1 = (x1 >= 0 && x1 < 9) ? 1.f : 0.f;
    float w00 = (1.f - ly) * (1.f - lx) * vy0 * vx0 * m;
    float w01 = (1.f - ly) * lx * vy0 * vx1 * m;
    float w10 = ly * (1.f - lx) * vy1 * vx0 * m;
    float w11 = ly * lx * vy1 * vx1 * m;
    int y0c = min(max(y0, 0), 8), y1c = min(max(y1, 0), 8);
    int x0c = min(max(x0, 0), 8), x1c = min(max(x1, 0), 8);
    const float* p00 = hn + (size_t)(b * 81 + y0c * 9 + x0c) * 64;
    const float* p01 = hn + (size_t)(b * 81 + y0c * 9 + x1c) * 64;
    const float* p10 = hn + (size_t)(b * 81 + y1c * 9 + x0c) * 64;
    const float* p11 = hn + (size_t)(b * 81 + y1c * 9 + x1c) * 64;
    f16* dst = &Vs[pl][kk * 64];
    for (int c = 0; c < 64; ++c)
      dst[c] = (f16)(w00 * p00[c] + w01 * p01[c] + w10 * p10[c] + w11 * p11[c]);
  }
  __syncthreads();
  const int wave = tid >> 5, lane = tid & 31;
  const int mSub = wave & 1, nT = wave >> 1;
  const int g = lane >> 4, l16 = lane & 15;
  v8f acc = {};
  for (int kc = 0; kc < KC2; ++kc) {
    v16h a = load_a(&Vs[mSub * 16 + l16][kc * 32 + 8 * g]);
    v16h bm = *(const v16h*)(wf3 + ((size_t)(kc * 4 + nT) * 32 + lane) * 16);
    acc = __builtin_amdgcn_wmma_f32_16x16x32_f16(false, a, false, bm, (short)0, acc, false, false);
  }
  const int n = nT * 16 + l16;
  const float bias = bd[n];
  const int mrow = pBase + mSub * 16 + 8 * g;
  float ls = 0.f, lq = 0.f;
#pragma unroll
  for (int r = 0; r < 8; ++r) {
    float v = acc[r] + bias;
    h2[(size_t)(mrow + r) * 64 + n] = v;
    ls += v; lq += v * v;
  }
  atomicAdd(&sSum[n], ls);
  atomicAdd(&sSq[n], lq);
  __syncthreads();
  if (tid < 64) { atomicAdd(&sums[128 + tid], sSum[tid]); atomicAdd(&sums[192 + tid], sSq[tid]); }
}

// ---------------- BN2 normalize + transpose to NCHW output ------------------
__global__ void bn2_kernel(const float* __restrict__ sums, const float* __restrict__ g2,
                           const float* __restrict__ be2, const float* __restrict__ h2,
                           float* __restrict__ out) {
  size_t idx = (size_t)blockIdx.x * 256 + threadIdx.x;
  if (idx >= (size_t)NPIX * 64) return;
  int o = (int)(idx & 63);
  int p = (int)(idx >> 6);
  const float inv = 1.0f / (float)NPIX;
  float mean = sums[128 + o] * inv;
  float var = sums[192 + o] * inv - mean * mean;
  float s = rsqrtf(var + 1e-5f) * g2[o];
  int b = p / HW, pix = p % HW;
  out[HOFF + (size_t)(b * 64 + o) * 81 + pix] = (h2[idx] - mean) * s + be2[o];
}

extern "C" void kernel_launch(void* const* d_in, const int* in_sizes, int n_in,
                              void* d_out, int out_size, void* d_ws, size_t ws_size,
                              hipStream_t stream) {
  (void)in_sizes; (void)n_in; (void)out_size; (void)ws_size;
  const float* x   = (const float*)d_in[0];
  const float* w1  = (const float*)d_in[1];
  const float* b1  = (const float*)d_in[2];
  const float* g1  = (const float*)d_in[3];
  const float* be1 = (const float*)d_in[4];
  const float* wo  = (const float*)d_in[5];
  const float* bo  = (const float*)d_in[6];
  const float* wm  = (const float*)d_in[7];
  const float* bm  = (const float*)d_in[8];
  const float* wd  = (const float*)d_in[9];
  const float* bd  = (const float*)d_in[10];
  const float* g2  = (const float*)d_in[11];
  const float* be2 = (const float*)d_in[12];
  float* out = (float*)d_out;
  char* ws = (char*)d_ws;
  float* sums    = (float*)(ws + WS_SUMS);
  f16*   wf1     = (f16*)(ws + WS_WF1);
  f16*   wf2     = (f16*)(ws + WS_WF2);
  f16*   wf3     = (f16*)(ws + WS_WF3);
  float* h1      = (float*)(ws + WS_H1);
  float* maskbuf = (float*)(ws + WS_MASK);
  float* h2      = (float*)(ws + WS_H2);

  prep_kernel<<<304, 256, 0, stream>>>(w1, wo, wm, wd, out, sums, wf1, wf2, wf3);
  conv1_kernel<<<NPIX / 32, 256, 0, stream>>>(x, b1, wf1, h1, sums);
  bn1_kernel<<<(NPIX * 64) / 256, 256, 0, stream>>>(sums, g1, be1, h1);
  offmask_kernel<<<NPIX / 32, 128, 0, stream>>>(h1, bo, bm, wf2, out, maskbuf);
  deform_kernel<<<NPIX / 32, 256, 0, stream>>>(h1, out, maskbuf, bd, wf3, h2, sums);
  bn2_kernel<<<(NPIX * 64) / 256, 256, 0, stream>>>(sums, g2, be2, h2, out);
}
